// TextGAT_29583734734921
// MI455X (gfx1250) — compile-verified, matrix operand order checked by reading, not run on hardware
//
#include <hip/hip_runtime.h>
#include <hip/hip_bf16.h>
#include <math.h>

// ---- shapes (fixed by the reference) ----
#define DIN   512    // input feature dim
#define HC    256    // H*C of layer-1 output
#define C1    128    // channels per head, layer 1
#define NCLS_ 4
#define NEG_SLOPE 0.2f
#define LDSPITCH 520 // 512 + 8 bf16 pad: row stride 1040B (16B aligned, bank-skewed)

typedef __attribute__((ext_vector_type(16))) __bf16 v16bf;
typedef __attribute__((ext_vector_type(8)))  __bf16 v8bf;
typedef __attribute__((ext_vector_type(8)))  float  v8f;

// ---- monotonic float<->uint encoding for atomicMax on floats ----
__device__ __forceinline__ unsigned fenc(float f) {
    unsigned u = __float_as_uint(f);
    return (u & 0x80000000u) ? ~u : (u | 0x80000000u);
}
__device__ __forceinline__ float fdec(unsigned u) {
    u = (u & 0x80000000u) ? (u & 0x7fffffffu) : ~u;
    return __uint_as_float(u);
}
__device__ __forceinline__ float lrelu(float v) { return v > 0.f ? v : NEG_SLOPE * v; }

// ---- W1 [512,256] f32 row-major  ->  Wt [256,512] bf16 (transposed) ----
__global__ void prep_w1_kernel(const float* __restrict__ W1, __bf16* __restrict__ Wt) {
    int idx = blockIdx.x * blockDim.x + threadIdx.x;
    if (idx >= HC * DIN) return;
    int n = idx / DIN, k = idx % DIN;
    Wt[idx] = (__bf16)W1[k * HC + n];
}

// ---- GEMM1: H1[N,256] = X[N,512] @ W1, bf16 WMMA, f32 accumulate ----
// one block per 16-row strip; A tile staged once in LDS (bf16);
// 8 waves x 2 N-tiles each cover all 256 output columns.
__global__ __launch_bounds__(256) void gemm1_wmma_kernel(
        const float* __restrict__ X, const __bf16* __restrict__ Wt,
        float* __restrict__ H1) {
    __shared__ __bf16 Alds[16 * LDSPITCH];   // 16.6 KB

    const int m0 = blockIdx.x * 16;

    // cooperative stage: 16 rows x 512 cols f32 -> bf16 in LDS (x read ONCE)
    for (int i = threadIdx.x; i < 16 * (DIN / 4); i += 256) {
        int r  = i >> 7;          // / (512/4)
        int c4 = i & 127;
        float4 f = ((const float4*)(X + (size_t)(m0 + r) * DIN))[c4];
        __bf16 t4[4];
        t4[0] = (__bf16)f.x; t4[1] = (__bf16)f.y;
        t4[2] = (__bf16)f.z; t4[3] = (__bf16)f.w;
        *(uint2*)&Alds[r * LDSPITCH + c4 * 4] = *(const uint2*)t4;   // 8B LDS store
    }
    __syncthreads();

    const int wave = threadIdx.x >> 5;
    const int lane = threadIdx.x & 31;
    const int half = lane >> 4;
    const int arow = lane & 15;

    v8f acc0 = {}, acc1 = {};
    const int col0 = (wave * 2 + 0) * 16 + (lane & 15);
    const int col1 = (wave * 2 + 1) * 16 + (lane & 15);
    const __bf16* wr0 = Wt + (size_t)col0 * DIN;
    const __bf16* wr1 = Wt + (size_t)col1 * DIN;

    for (int k0 = 0; k0 < DIN; k0 += 32) {
        // A fragment from LDS: lane holds K in [k0+8h, +8) and [k0+16+8h, +8)
        v8bf a0 = *(const v8bf*)&Alds[arow * LDSPITCH + k0 + 8 * half];
        v8bf a1 = *(const v8bf*)&Alds[arow * LDSPITCH + k0 + 16 + 8 * half];
        v16bf a;
#pragma unroll
        for (int e = 0; e < 8; ++e) { a[e] = a0[e]; a[8 + e] = a1[e]; }

        // B fragments from L2-resident Wt (bf16, K-contiguous per column)
        v8bf b00 = *(const v8bf*)(wr0 + k0 + 8 * half);
        v8bf b01 = *(const v8bf*)(wr0 + k0 + 16 + 8 * half);
        v16bf b;
#pragma unroll
        for (int e = 0; e < 8; ++e) { b[e] = b00[e]; b[8 + e] = b01[e]; }
        acc0 = __builtin_amdgcn_wmma_f32_16x16x32_bf16(
                false, a, false, b, (short)0, acc0, false, false);

        v8bf b10 = *(const v8bf*)(wr1 + k0 + 8 * half);
        v8bf b11 = *(const v8bf*)(wr1 + k0 + 16 + 8 * half);
        v16bf b2;
#pragma unroll
        for (int e = 0; e < 8; ++e) { b2[e] = b10[e]; b2[8 + e] = b11[e]; }
        acc1 = __builtin_amdgcn_wmma_f32_16x16x32_bf16(
                false, a, false, b2, (short)0, acc1, false, false);
    }

    // D layout: vgpr r -> M = m0 + r + 8*half, N = col
#pragma unroll
    for (int r = 0; r < 8; ++r) {
        H1[(size_t)(m0 + r + 8 * half) * HC + col0] = acc0[r];
        H1[(size_t)(m0 + r + 8 * half) * HC + col1] = acc1[r];
    }
}

// ---- alpha_src1/alpha_dst1 [N,2]: per (node, head) dot over 128 channels ----
__global__ void alpha1_kernel(const float* __restrict__ H1,
                              const float* __restrict__ asv, const float* __restrict__ adv,
                              float* __restrict__ AS1, float* __restrict__ AD1, int Nn) {
    int t = blockIdx.x * blockDim.x + threadIdx.x;
    if (t >= Nn * 2) return;
    int n = t >> 1, hd = t & 1;
    const float* hp = H1 + (size_t)n * HC + hd * C1;
    const float* ap = asv + hd * C1;
    const float* bp = adv + hd * C1;
    float sa = 0.f, sb = 0.f;
#pragma unroll 4
    for (int c = 0; c < C1; ++c) { float v = hp[c]; sa += v * ap[c]; sb += v * bp[c]; }
    AS1[t] = sa; AD1[t] = sb;
}

// ---- zero out1, reset m/z for layer 1 ----
__global__ void init1_kernel(float* __restrict__ OUT1, unsigned* __restrict__ M1,
                             float* __restrict__ Z1, int Nn) {
    int idx = blockIdx.x * blockDim.x + threadIdx.x;
    if (idx < Nn * HC) OUT1[idx] = 0.f;
    if (idx < Nn * 2) { M1[idx] = 0u; Z1[idx] = 0.f; }
}

// ---- edge pass A: segment max of leaky_relu(as[src]+ad[dst]) per head ----
__global__ void edge_max_kernel(const int* __restrict__ src, const int* __restrict__ dst,
                                int E_, int nE, const float* __restrict__ AS,
                                const float* __restrict__ AD, unsigned* __restrict__ M,
                                int Hh) {
    int e = blockIdx.x * blockDim.x + threadIdx.x;
    if (e >= nE) return;
    int s, d;
    if (e < E_) { s = src[e]; d = dst[e]; } else { s = d = e - E_; }
    for (int h = 0; h < Hh; ++h) {
        float l = lrelu(AS[s * Hh + h] + AD[d * Hh + h]);
        atomicMax(&M[d * Hh + h], fenc(l));
    }
}

// ---- edge pass B: z[dst] += exp(l - m[dst]) ----
__global__ void edge_sum_kernel(const int* __restrict__ src, const int* __restrict__ dst,
                                int E_, int nE, const float* __restrict__ AS,
                                const float* __restrict__ AD, const unsigned* __restrict__ M,
                                float* __restrict__ Z, int Hh) {
    int e = blockIdx.x * blockDim.x + threadIdx.x;
    if (e >= nE) return;
    int s, d;
    if (e < E_) { s = src[e]; d = dst[e]; } else { s = d = e - E_; }
    for (int h = 0; h < Hh; ++h) {
        float l = lrelu(AS[s * Hh + h] + AD[d * Hh + h]);
        atomicAdd(&Z[d * Hh + h], __expf(l - fdec(M[d * Hh + h])));
    }
}

// ---- edge pass C (layer 1): out1[dst] += h1[src] * w ; block per edge ----
__global__ __launch_bounds__(256) void edge_aggr1_kernel(
        const int* __restrict__ src, const int* __restrict__ dst, int E_, int nE,
        const float* __restrict__ AS, const float* __restrict__ AD,
        const unsigned* __restrict__ M, const float* __restrict__ Z,
        const float* __restrict__ H1, float* __restrict__ OUT1) {
    int e = blockIdx.x;
    if (e >= nE) return;
    int s, d;
    if (e < E_) { s = src[e]; d = dst[e]; } else { s = d = e - E_; }
    int t = threadIdx.x;        // 0..255 flat channel
    int h = t >> 7;             // head
    float l = lrelu(AS[s * 2 + h] + AD[d * 2 + h]);
    float w = __expf(l - fdec(M[d * 2 + h])) / Z[d * 2 + h];
    atomicAdd(&OUT1[(size_t)d * HC + t], H1[(size_t)s * HC + t] * w);
}

// ---- x2 = elu(out1 + b1), in place ----
__global__ void elu_bias1_kernel(float* __restrict__ OUT1, const float* __restrict__ b1, int Nn) {
    int idx = blockIdx.x * blockDim.x + threadIdx.x;
    if (idx >= Nn * HC) return;
    float v = OUT1[idx] + b1[idx % HC];
    OUT1[idx] = v > 0.f ? v : __expf(v) - 1.f;
}

// ---- layer 2 linear + attention coefficients + reset m2/z2 ----
__global__ void gemm2_kernel(const float* __restrict__ X2, const float* __restrict__ W2,
                             const float* __restrict__ as2v, const float* __restrict__ ad2v,
                             float* __restrict__ H2, float* __restrict__ AS2,
                             float* __restrict__ AD2, unsigned* __restrict__ M2,
                             float* __restrict__ Z2, int Nn) {
    int n = blockIdx.x * blockDim.x + threadIdx.x;
    if (n >= Nn) return;
    const float* xr = X2 + (size_t)n * HC;
    float4 acc = {0.f, 0.f, 0.f, 0.f};
    for (int k = 0; k < HC; ++k) {
        float  xv = xr[k];
        float4 wr = ((const float4*)W2)[k];
        acc.x += xv * wr.x; acc.y += xv * wr.y; acc.z += xv * wr.z; acc.w += xv * wr.w;
    }
    ((float4*)H2)[n] = acc;
    AS2[n] = acc.x * as2v[0] + acc.y * as2v[1] + acc.z * as2v[2] + acc.w * as2v[3];
    AD2[n] = acc.x * ad2v[0] + acc.y * ad2v[1] + acc.z * ad2v[2] + acc.w * ad2v[3];
    M2[n] = 0u; Z2[n] = 0.f;
}

// ---- out = b2 (bias folded into the scatter-add init) ----
__global__ void init_out2_kernel(float* __restrict__ OUT, const float* __restrict__ b2, int Nn) {
    int idx = blockIdx.x * blockDim.x + threadIdx.x;
    if (idx >= Nn * NCLS_) return;
    OUT[idx] = b2[idx & 3];
}

// ---- edge pass C (layer 2): out[dst] += h2[src] * w ; thread per edge ----
__global__ void edge_aggr2_kernel(const int* __restrict__ src, const int* __restrict__ dst,
                                  int E_, int nE, const float* __restrict__ AS,
                                  const float* __restrict__ AD, const unsigned* __restrict__ M,
                                  const float* __restrict__ Z, const float* __restrict__ H2,
                                  float* __restrict__ OUT) {
    int e = blockIdx.x * blockDim.x + threadIdx.x;
    if (e >= nE) return;
    int s, d;
    if (e < E_) { s = src[e]; d = dst[e]; } else { s = d = e - E_; }
    float l = lrelu(AS[s] + AD[d]);
    float w = __expf(l - fdec(M[d])) / Z[d];
    float4 hv = ((const float4*)H2)[s];
    atomicAdd(&OUT[d * 4 + 0], hv.x * w);
    atomicAdd(&OUT[d * 4 + 1], hv.y * w);
    atomicAdd(&OUT[d * 4 + 2], hv.z * w);
    atomicAdd(&OUT[d * 4 + 3], hv.w * w);
}

extern "C" void kernel_launch(void* const* d_in, const int* in_sizes, int n_in,
                              void* d_out, int out_size, void* d_ws, size_t ws_size,
                              hipStream_t stream) {
    const float* x      = (const float*)d_in[0];
    const int*   eidx   = (const int*)  d_in[1];
    const float* W1     = (const float*)d_in[2];
    const float* a_src1 = (const float*)d_in[3];
    const float* a_dst1 = (const float*)d_in[4];
    const float* b1     = (const float*)d_in[5];
    const float* W2     = (const float*)d_in[6];
    const float* a_src2 = (const float*)d_in[7];
    const float* a_dst2 = (const float*)d_in[8];
    const float* b2     = (const float*)d_in[9];
    float* out = (float*)d_out;

    const int Nn = in_sizes[0] / DIN;   // 50000
    const int E_ = in_sizes[1] / 2;     // 800000
    const int*  src = eidx;
    const int*  dst = eidx + E_;
    const int   nE  = E_ + Nn;          // with self loops

    // ---- carve workspace ----
    char* p = (char*)d_ws;
    auto carve = [&](size_t bytes) -> char* {
        char* r = p; p += (bytes + 255) & ~(size_t)255; return r;
    };
    float*    H1   = (float*)   carve((size_t)Nn * HC * sizeof(float));
    float*    OUT1 = (float*)   carve((size_t)Nn * HC * sizeof(float));
    __bf16*   Wt   = (__bf16*)  carve((size_t)HC * DIN * sizeof(__bf16));
    float*    AS1  = (float*)   carve((size_t)Nn * 2 * sizeof(float));
    float*    AD1  = (float*)   carve((size_t)Nn * 2 * sizeof(float));
    unsigned* M1   = (unsigned*)carve((size_t)Nn * 2 * sizeof(unsigned));
    float*    Z1   = (float*)   carve((size_t)Nn * 2 * sizeof(float));
    float*    H2   = (float*)   carve((size_t)Nn * NCLS_ * sizeof(float));
    float*    AS2  = (float*)   carve((size_t)Nn * sizeof(float));
    float*    AD2  = (float*)   carve((size_t)Nn * sizeof(float));
    unsigned* M2   = (unsigned*)carve((size_t)Nn * sizeof(unsigned));
    float*    Z2   = (float*)   carve((size_t)Nn * sizeof(float));
    (void)ws_size; (void)n_in; (void)out_size;

    const int B = 256;
    // layer 1
    prep_w1_kernel<<<(HC * DIN + B - 1) / B, B, 0, stream>>>(W1, Wt);
    gemm1_wmma_kernel<<<Nn / 16, B, 0, stream>>>(x, Wt, H1);
    alpha1_kernel<<<(Nn * 2 + B - 1) / B, B, 0, stream>>>(H1, a_src1, a_dst1, AS1, AD1, Nn);
    init1_kernel<<<((size_t)Nn * HC + B - 1) / B, B, 0, stream>>>(OUT1, M1, Z1, Nn);
    edge_max_kernel<<<(nE + B - 1) / B, B, 0, stream>>>(src, dst, E_, nE, AS1, AD1, M1, 2);
    edge_sum_kernel<<<(nE + B - 1) / B, B, 0, stream>>>(src, dst, E_, nE, AS1, AD1, M1, Z1, 2);
    edge_aggr1_kernel<<<nE, B, 0, stream>>>(src, dst, E_, nE, AS1, AD1, M1, Z1, H1, OUT1);
    elu_bias1_kernel<<<((size_t)Nn * HC + B - 1) / B, B, 0, stream>>>(OUT1, b1, Nn);
    // layer 2
    gemm2_kernel<<<(Nn + B - 1) / B, B, 0, stream>>>(OUT1, W2, a_src2, a_dst2,
                                                     H2, AS2, AD2, M2, Z2, Nn);
    init_out2_kernel<<<(Nn * NCLS_ + B - 1) / B, B, 0, stream>>>(out, b2, Nn);
    edge_max_kernel<<<(nE + B - 1) / B, B, 0, stream>>>(src, dst, E_, nE, AS2, AD2, M2, 1);
    edge_sum_kernel<<<(nE + B - 1) / B, B, 0, stream>>>(src, dst, E_, nE, AS2, AD2, M2, Z2, 1);
    edge_aggr2_kernel<<<(nE + B - 1) / B, B, 0, stream>>>(src, dst, E_, nE, AS2, AD2, M2, Z2, H2, out);
}
